// SortNet_48112223650043
// MI455X (gfx1250) — matrix operand order repaired and checked by reading, not compiled, not run on hardware
//
#include <hip/hip_runtime.h>
#include <hip/hip_bf16.h>

typedef _Float16 f16;
typedef __attribute__((ext_vector_type(16))) _Float16 v16h;
typedef __attribute__((ext_vector_type(8)))  float    v8f;
typedef __attribute__((ext_vector_type(4)))  int      v4i;

#define NEG_INF (-3.402823466e38f)

// ---------------------------------------------------------------------------
// Optional gfx1250 async global->LDS path (guarded; falls back to plain copy)
// Probe-learned signature: (AS1 v4i*, AS3 v4i*, imm offset, imm cpol)
// ---------------------------------------------------------------------------
#if defined(__has_builtin)
#  if __has_builtin(__builtin_amdgcn_global_load_async_to_lds_b128) && \
      __has_builtin(__builtin_amdgcn_s_wait_asynccnt)
#    define HAVE_ASYNC_LDS 1
#  endif
#endif

#ifdef HAVE_ASYNC_LDS
typedef __attribute__((address_space(1))) v4i gv4i;
typedef __attribute__((address_space(3))) v4i lv4i;
// Generic LDS address low 32 bits are the LDS offset (ISA 10.2 aperture rule),
// so integer casts give correctly-typed as1/as3 pointers.
__device__ __forceinline__ void async_copy_b128(const void* gsrc, void* ldst) {
  __builtin_amdgcn_global_load_async_to_lds_b128(
      (gv4i*)(unsigned long long)gsrc,
      (lv4i*)(unsigned)(unsigned long long)ldst, 0, 0);
}
#endif

// ---------------------------------------------------------------------------
// WMMA helpers (CDNA5 wave32, V_WMMA_F32_16X16X32_F16)
// ---------------------------------------------------------------------------
__device__ __forceinline__ v8f wmma_f32_16x16x32(v16h a, v16h b, v8f c) {
  return __builtin_amdgcn_wmma_f32_16x16x32_f16(false, a, false, b, (short)0, c,
                                                false, false);
}

// A-matrix 16x32 f16 fragment from LDS, row stride lda (ISA 7.12.2 layout).
// Two contiguous 16B groups per lane -> 2x ds_load_b128.
__device__ __forceinline__ v16h load_a_frag(const f16* A, int lda) {
  int lane = threadIdx.x & 31;
  int row  = lane & 15;
  int koff = (lane >> 4) << 3;  // 0 or 8
  const f16* p = A + row * lda + koff;
  v16h a;
#pragma unroll
  for (int h = 0; h < 8; ++h) a[h] = p[h];           // K = koff + 0..7
#pragma unroll
  for (int h = 0; h < 8; ++h) a[8 + h] = p[16 + h];  // K = koff + 16..23
  return a;
}

// B-matrix 32x16 f16 fragment from a pre-swizzled weight buffer: each lane's
// 16 values are contiguous (32B aligned) -> 2x global_load_b128.
__device__ __forceinline__ v16h load_b_frag_swz(const f16* Wsw, int tile) {
  int lane = threadIdx.x & 31;
  return *(const v16h*)(Wsw + ((size_t)tile << 9) + (lane << 4));
}

// D-matrix f32 16x16: VGPR j holds row M = j + 8*(lane>=16), col = lane%16.

// ---------------------------------------------------------------------------
// Weight prep: f32 -> f16, zero-pad to (drows,dcols), store in B-fragment
// swizzled order.  (k,n) -> [ (k/32)*(dcols/16)+(n/16) ][ (n&15)|((k>>3&1)<<4) ][ (k&7)|((k>>4&1)<<3) ]
// Requires drows % 32 == 0, dcols % 16 == 0.
// ---------------------------------------------------------------------------
__global__ void convf16_swz(const float* __restrict__ src, f16* __restrict__ dst,
                            int srows, int scols, int drows, int dcols) {
  int i = blockIdx.x * 256 + threadIdx.x;
  if (i >= drows * dcols) return;
  int r = i / dcols, c = i % dcols;
  float v = (r < srows && c < scols) ? src[r * scols + c] : 0.f;
  int ti   = (r >> 5) * (dcols >> 4) + (c >> 4);
  int lane = (c & 15) | (((r >> 3) & 1) << 4);
  int h    = (r & 7) | (((r >> 4) & 1) << 3);
  dst[((size_t)ti << 9) + (lane << 4) + h] = (f16)v;
}

__global__ void padbias_f32(const float* __restrict__ src, float* __restrict__ dst,
                            int n, int npad) {
  int i = blockIdx.x * 256 + threadIdx.x;
  if (i < npad) dst[i] = (i < n) ? src[i] : 0.f;
}

// ---------------------------------------------------------------------------
// Generic tiled GEMM:  C[M,N] = act(A[M,K] @ Bswz[K,N] + bias)
// Block = 128 threads (4 waves), block tile 64(M) x 64(N), K step 32.
// A staged in LDS (f32 converted, f16 async-copied if available); B fragments
// come straight from the swizzled global weights (L2-resident).
// ---------------------------------------------------------------------------
#define GA_LD 40  // 32 + 8 pad (80B rows, 16B aligned)

template <typename AT, bool RELU, bool OUTF16>
__global__ __launch_bounds__(128) void gemm_tiled(
    const AT* __restrict__ A, const f16* __restrict__ Bsw,
    const float* __restrict__ bias, void* __restrict__ Cout,
    int M, int Kd, int N) {
  __shared__ f16 As[64 * GA_LD];
  const int tid  = threadIdx.x;
  const int wave = tid >> 5;
  const int m0   = blockIdx.x * 64;
  const int n0   = blockIdx.y * 64;
  const int ntile_row = N >> 4;  // B tiles per 32-K slab

  v8f acc[4] = {v8f{}, v8f{}, v8f{}, v8f{}};

  for (int kc = 0; kc < Kd; kc += 32) {
    {  // stage A tile 64x32
      int r = tid >> 1, c0 = (tid & 1) * 16;
      const AT* src = A + (size_t)(m0 + r) * Kd + kc + c0;
      f16* dst = &As[r * GA_LD + c0];
      if constexpr (sizeof(AT) == 2) {
#ifdef HAVE_ASYNC_LDS
        async_copy_b128(src, dst);          // 16B
        async_copy_b128(src + 8, dst + 8);  // 16B
#else
#pragma unroll
        for (int i = 0; i < 16; ++i) dst[i] = (f16)src[i];
#endif
      } else {
#pragma unroll
        for (int i = 0; i < 16; ++i) dst[i] = (f16)src[i];
        if (kc + 32 < Kd) __builtin_prefetch(src + 32, 0, 1);  // global_prefetch_b8
      }
    }
#ifdef HAVE_ASYNC_LDS
    if constexpr (sizeof(AT) == 2) __builtin_amdgcn_s_wait_asynccnt(0);
#endif
    __syncthreads();
    v16h a = load_a_frag(&As[(wave * 16) * GA_LD], GA_LD);
    int tbase = (kc >> 5) * ntile_row + (n0 >> 4);
#pragma unroll
    for (int nt = 0; nt < 4; ++nt) {
      v16h b = load_b_frag_swz(Bsw, tbase + nt);
      acc[nt] = wmma_f32_16x16x32(a, b, acc[nt]);
    }
    __syncthreads();
  }

  const int lane = tid & 31;
  const int col = lane & 15, rbase = (lane >> 4) << 3;
  const int grow = m0 + wave * 16 + rbase;
#pragma unroll
  for (int nt = 0; nt < 4; ++nt) {
    int gcol = n0 + nt * 16 + col;
    float bv = bias ? bias[gcol] : 0.f;
#pragma unroll
    for (int j = 0; j < 8; ++j) {
      float v = acc[nt][j] + bv;
      if (RELU) v = fmaxf(v, 0.f);
      if (OUTF16)
        ((f16*)Cout)[(size_t)(grow + j) * N + gcol] = (f16)v;
      else
        ((float*)Cout)[(size_t)(grow + j) * N + gcol] = v;
    }
  }
}

// ---------------------------------------------------------------------------
// score[i] = H2[i,:] . w2 + b2
// ---------------------------------------------------------------------------
__global__ void score_kernel(const f16* __restrict__ H2,
                             const float* __restrict__ w2,
                             const float* __restrict__ b2,
                             float* __restrict__ score, int M) {
  int i = blockIdx.x * 256 + threadIdx.x;
  if (i >= M) return;
  float s = b2[0];
  const f16* r = H2 + (size_t)i * 64;
#pragma unroll
  for (int k = 0; k < 64; ++k) s += (float)r[k] * w2[k];
  score[i] = s;
}

// ---------------------------------------------------------------------------
// top-64 per batch (iterative argmax, lower index wins ties)
// ---------------------------------------------------------------------------
__global__ __launch_bounds__(256) void topk_kernel(
    const float* __restrict__ score, float* __restrict__ scratch,
    int* __restrict__ topk, float* __restrict__ newxyz,
    const float* __restrict__ points) {
  int b = blockIdx.x, t = threadIdx.x;
  float* s = scratch + (size_t)b * 4096;
  for (int i = t; i < 4096; i += 256) s[i] = score[(size_t)b * 4096 + i];
  __syncthreads();
  __shared__ float bv[256];
  __shared__ int bi[256];
  for (int k = 0; k < 64; ++k) {
    float best = NEG_INF;
    int bidx = 0x7fffffff;
    for (int i = t; i < 4096; i += 256) {
      float v = s[i];
      if (v > best || (v == best && i < bidx)) { best = v; bidx = i; }
    }
    bv[t] = best; bi[t] = bidx;
    __syncthreads();
    for (int off = 128; off > 0; off >>= 1) {
      if (t < off) {
        if (bv[t + off] > bv[t] || (bv[t + off] == bv[t] && bi[t + off] < bi[t])) {
          bv[t] = bv[t + off]; bi[t] = bi[t + off];
        }
      }
      __syncthreads();
    }
    if (t == 0) {
      int idx = bi[0];
      topk[b * 64 + k] = idx;
      s[idx] = NEG_INF;
      const float* pp = points + ((size_t)b * 4096 + idx) * 6;
      float* nx = newxyz + (size_t)(b * 64 + k) * 3;
      nx[0] = pp[0]; nx[1] = pp[1]; nx[2] = pp[2];
    }
    __syncthreads();
  }
}

// ---------------------------------------------------------------------------
// Ball query (wave32 ballot keeps ascending-index order; pad with first hit)
// ---------------------------------------------------------------------------
template <int NS>
__global__ __launch_bounds__(32) void ball_kernel(
    const float* __restrict__ points, const float* __restrict__ newxyz,
    int* __restrict__ out, float r2) {
  int bs = blockIdx.x;
  int b = bs >> 6;
  int lane = threadIdx.x;
  float cx = newxyz[bs * 3 + 0], cy = newxyz[bs * 3 + 1], cz = newxyz[bs * 3 + 2];
  int* dst = out + (size_t)bs * NS;
  int count = 0, first = -1;
  for (int base = 0; base < 4096 && count < NS; base += 32) {
    int p = base + lane;
    const float* pp = points + ((size_t)b * 4096 + p) * 6;
    float dx = pp[0] - cx, dy = pp[1] - cy, dz = pp[2] - cz;
    bool flag = (dx * dx + dy * dy + dz * dz) <= r2;
    unsigned mask = (unsigned)__ballot(flag);
    int prefix = __popc(mask & ((1u << lane) - 1u));
    int pos = count + prefix;
    if (flag && pos < NS) dst[pos] = p;
    if (first < 0 && mask) first = base + __ffs((int)mask) - 1;
    count += __popc(mask);
  }
  for (int q = count + lane; q < NS; q += 32) dst[q] = first;
}

// ---------------------------------------------------------------------------
// SA layer helpers (all-relu MLP on a 32-row LDS tile; B from swizzled global)
// ---------------------------------------------------------------------------
template <int KD, int ND, int ALD>
__device__ __forceinline__ void mlp_layer_store(
    const f16* __restrict__ As, const f16* __restrict__ Wsw,
    const float* __restrict__ bg, f16* __restrict__ Hout, int hld) {
  constexpr int T = 2 * (ND / 16);
  constexpr int TPW = T / 4;
  const int wave = threadIdx.x >> 5;
  v8f acc[TPW];
#pragma unroll
  for (int i = 0; i < TPW; ++i) acc[i] = v8f{};
  for (int kc = 0; kc < KD; kc += 32) {
#pragma unroll
    for (int i = 0; i < TPW; ++i) {
      int t = wave + i * 4;
      int mt = t & 1, nt = t >> 1;
      v16h a = load_a_frag(As + mt * 16 * ALD + kc, ALD);
      v16h b = load_b_frag_swz(Wsw, (kc >> 5) * (ND >> 4) + nt);
      acc[i] = wmma_f32_16x16x32(a, b, acc[i]);
    }
  }
  const int lane = threadIdx.x & 31;
  const int col = lane & 15, rbase = (lane >> 4) << 3;
#pragma unroll
  for (int i = 0; i < TPW; ++i) {
    int t = wave + i * 4;
    int mt = t & 1, nt = t >> 1;
    float bv = bg[nt * 16 + col];
#pragma unroll
    for (int j = 0; j < 8; ++j) {
      float v = fmaxf(acc[i][j] + bv, 0.f);
      Hout[(mt * 16 + rbase + j) * hld + nt * 16 + col] = (f16)v;
    }
  }
}

template <int KD, int ND, int ALD>
__device__ __forceinline__ void mlp_layer_max(
    const f16* __restrict__ As, const f16* __restrict__ Wsw,
    const float* __restrict__ bg, int* __restrict__ maxbits) {
  constexpr int T = 2 * (ND / 16);
  constexpr int TPW = T / 4;
  const int wave = threadIdx.x >> 5;
  v8f acc[TPW];
#pragma unroll
  for (int i = 0; i < TPW; ++i) acc[i] = v8f{};
  for (int kc = 0; kc < KD; kc += 32) {
#pragma unroll
    for (int i = 0; i < TPW; ++i) {
      int t = wave + i * 4;
      int mt = t & 1, nt = t >> 1;
      v16h a = load_a_frag(As + mt * 16 * ALD + kc, ALD);
      v16h b = load_b_frag_swz(Wsw, (kc >> 5) * (ND >> 4) + nt);
      acc[i] = wmma_f32_16x16x32(a, b, acc[i]);
    }
  }
  const int lane = threadIdx.x & 31;
  const int col = lane & 15;
#pragma unroll
  for (int i = 0; i < TPW; ++i) {
    int t = wave + i * 4;
    int nt = t >> 1;
    float bv = bg[nt * 16 + col];
    float mv = 0.f;  // relu output >= 0
#pragma unroll
    for (int j = 0; j < 8; ++j) mv = fmaxf(mv, fmaxf(acc[i][j] + bv, 0.f));
    atomicMax(&maxbits[nt * 16 + col], __float_as_int(mv));  // non-neg floats
  }
}

// ---------------------------------------------------------------------------
// SA kernel: one block (128 thr) per (batch, center); 32-row grouped chunks.
// ---------------------------------------------------------------------------
#define K0PAD 544
#define G_LD 552  // 544 + 8 (1104B rows, 16B aligned)

template <int NS, int M0, int M1, int M2>
__global__ __launch_bounds__(128) void sa_kernel(
    const float* __restrict__ points, const float* __restrict__ feats,
    const int* __restrict__ ballidx, const float* __restrict__ newxyz,
    const f16* __restrict__ W0, const float* __restrict__ b0,
    const f16* __restrict__ W1, const float* __restrict__ b1,
    const f16* __restrict__ W2, const float* __restrict__ b2,
    float* __restrict__ sa_out, int out_off) {
  constexpr int H0LD = M0 + 8;
  constexpr int H1LD = M1 + 8;
  __shared__ f16 G[32 * G_LD];
  __shared__ f16 H0s[32 * H0LD];
  __shared__ f16 H1s[32 * H1LD];
  __shared__ int maxbits[M2];

  const int bs = blockIdx.x;
  const int b = bs >> 6;
  const int tid = threadIdx.x;

  if (tid < M2) maxbits[tid] = 0;
  const float cx = newxyz[bs * 3 + 0];
  const float cy = newxyz[bs * 3 + 1];
  const float cz = newxyz[bs * 3 + 2];
  __syncthreads();

  constexpr int NCHUNK = (NS + 31) / 32;
  for (int chunk = 0; chunk < NCHUNK; ++chunk) {
    {  // gather 32 rows; thread t: row = t/4, 128-feature-col segment t%4
      int r = tid >> 2, seg = tid & 3;
      int j = chunk * 32 + r;
      if (j >= NS) j = 0;  // duplicate row 0; harmless under max
      int pidx = ballidx[(size_t)bs * NS + j];
      const float* frow = feats + ((size_t)b * 4096 + pidx) * 512;
      f16* grow = &G[r * G_LD];
      const float4* f4 = (const float4*)(frow + seg * 128);
      f16* gw = grow + seg * 128;
#pragma unroll 8
      for (int q = 0; q < 32; ++q) {
        float4 v = f4[q];
        gw[4 * q + 0] = (f16)v.x;
        gw[4 * q + 1] = (f16)v.y;
        gw[4 * q + 2] = (f16)v.z;
        gw[4 * q + 3] = (f16)v.w;
      }
      if (seg == 3) {  // xyz-center + zero pad (cols 512..543)
        const float* prow = points + ((size_t)b * 4096 + pidx) * 6;
        grow[512] = (f16)(prow[0] - cx);
        grow[513] = (f16)(prow[1] - cy);
        grow[514] = (f16)(prow[2] - cz);
#pragma unroll
        for (int c = 515; c < 544; ++c) grow[c] = (f16)0.f;
      }
    }
    __syncthreads();
    mlp_layer_store<K0PAD, M0, G_LD>(G, W0, b0, H0s, H0LD);
    __syncthreads();
    mlp_layer_store<M0, M1, H0LD>(H0s, W1, b1, H1s, H1LD);
    __syncthreads();
    mlp_layer_max<M1, M2, H1LD>(H1s, W2, b2, maxbits);
    __syncthreads();
  }
  if (tid < M2)
    sa_out[(size_t)bs * 320 + out_off + tid] = __int_as_float(maxbits[tid]);
}

// ---------------------------------------------------------------------------
// Final assembly: out[b,k,:] = [agg(505) | score(1) | points(6)]
// ---------------------------------------------------------------------------
__global__ void assemble_kernel(const float* __restrict__ aggout,
                                const float* __restrict__ score,
                                const float* __restrict__ points,
                                const int* __restrict__ topk,
                                float* __restrict__ out) {
  int i = blockIdx.x * 256 + threadIdx.x;
  if (i >= 1024 * 512) return;
  int bs = i >> 9, c = i & 511, b = bs >> 6;
  float v;
  if (c < 505) v = aggout[(size_t)bs * 512 + c];
  else if (c == 505) v = score[(size_t)b * 4096 + topk[bs]];
  else v = points[((size_t)b * 4096 + topk[bs]) * 6 + (c - 506)];
  out[i] = v;
}

// ---------------------------------------------------------------------------
// Host launcher
// ---------------------------------------------------------------------------
extern "C" void kernel_launch(void* const* d_in, const int* in_sizes, int n_in,
                              void* d_out, int out_size, void* d_ws, size_t ws_size,
                              hipStream_t stream) {
  (void)in_sizes; (void)n_in; (void)out_size; (void)ws_size;
  const float* points   = (const float*)d_in[0];
  const float* features = (const float*)d_in[1];
  const float* fc_w0 = (const float*)d_in[2];  const float* fc_b0 = (const float*)d_in[3];
  const float* fc_w1 = (const float*)d_in[4];  const float* fc_b1 = (const float*)d_in[5];
  const float* fc_w2 = (const float*)d_in[6];  const float* fc_b2 = (const float*)d_in[7];
  const float* sw0[3] = {(const float*)d_in[8],  (const float*)d_in[14], (const float*)d_in[20]};
  const float* sb0[3] = {(const float*)d_in[9],  (const float*)d_in[15], (const float*)d_in[21]};
  const float* sw1[3] = {(const float*)d_in[10], (const float*)d_in[16], (const float*)d_in[22]};
  const float* sb1[3] = {(const float*)d_in[11], (const float*)d_in[17], (const float*)d_in[23]};
  const float* sw2[3] = {(const float*)d_in[12], (const float*)d_in[18], (const float*)d_in[24]};
  const float* sb2[3] = {(const float*)d_in[13], (const float*)d_in[19], (const float*)d_in[25]};
  const float* agg_w0 = (const float*)d_in[26]; const float* agg_b0 = (const float*)d_in[27];
  const float* agg_w1 = (const float*)d_in[28]; const float* agg_b1 = (const float*)d_in[29];
  const float* agg_w2 = (const float*)d_in[30]; const float* agg_b2 = (const float*)d_in[31];
  float* out = (float*)d_out;

  size_t off = 0;
  auto alloc = [&](size_t bytes) -> char* {
    char* p = (char*)d_ws + off;
    off += (bytes + 255) & ~(size_t)255;
    return p;
  };
  const int M0s[3] = {32, 64, 64}, M1s[3] = {32, 64, 96}, M2s[3] = {64, 128, 128};
  const int NSs[3] = {16, 32, 128};
  const float R2s[3] = {0.1f * 0.1f, 0.2f * 0.2f, 0.4f * 0.4f};
  const int OFFs[3] = {0, 64, 192};

  f16* fc0h  = (f16*)alloc((size_t)512 * 256 * 2);
  f16* fc1h  = (f16*)alloc((size_t)256 * 64 * 2);
  f16* w0h[3]; f16* w1h[3]; f16* w2h[3];
  for (int s = 0; s < 3; ++s) {
    w0h[s] = (f16*)alloc((size_t)K0PAD * M0s[s] * 2);
    w1h[s] = (f16*)alloc((size_t)M0s[s] * M1s[s] * 2);
    w2h[s] = (f16*)alloc((size_t)M1s[s] * M2s[s] * 2);
  }
  f16* agg0h = (f16*)alloc((size_t)320 * 256 * 2);
  f16* agg1h = (f16*)alloc((size_t)256 * 256 * 2);
  f16* agg2h = (f16*)alloc((size_t)256 * 512 * 2);
  float* aggb2p = (float*)alloc(512 * 4);
  f16* H1 = (f16*)alloc((size_t)65536 * 256 * 2);
  f16* H2 = (f16*)alloc((size_t)65536 * 64 * 2);
  float* score   = (float*)alloc((size_t)65536 * 4);
  float* scratch = (float*)alloc((size_t)65536 * 4);
  int*   topk    = (int*)alloc(1024 * 4);
  float* newxyz  = (float*)alloc((size_t)1024 * 3 * 4);
  int* ball[3];
  for (int s = 0; s < 3; ++s) ball[s] = (int*)alloc((size_t)1024 * NSs[s] * 4);
  float* sa_out = (float*)alloc((size_t)1024 * 320 * 4);
  f16* aggH0 = (f16*)alloc((size_t)1024 * 256 * 2);
  f16* aggH1 = (f16*)alloc((size_t)1024 * 256 * 2);
  float* aggout = (float*)alloc((size_t)1024 * 512 * 4);

  auto conv = [&](const float* src, f16* dst, int sr, int sc, int dr, int dc) {
    int n = dr * dc;
    convf16_swz<<<(n + 255) / 256, 256, 0, stream>>>(src, dst, sr, sc, dr, dc);
  };
  // --- weight prep (swizzled into WMMA B-fragment order) ---
  conv(fc_w0, fc0h, 512, 256, 512, 256);
  conv(fc_w1, fc1h, 256, 64, 256, 64);
  for (int s = 0; s < 3; ++s) {
    conv(sw0[s], w0h[s], 515, M0s[s], K0PAD, M0s[s]);
    conv(sw1[s], w1h[s], M0s[s], M1s[s], M0s[s], M1s[s]);
    conv(sw2[s], w2h[s], M1s[s], M2s[s], M1s[s], M2s[s]);
  }
  conv(agg_w0, agg0h, 320, 256, 320, 256);
  conv(agg_w1, agg1h, 256, 256, 256, 256);
  conv(agg_w2, agg2h, 256, 505, 256, 512);
  padbias_f32<<<2, 256, 0, stream>>>(agg_b2, aggb2p, 505, 512);

  // --- score MLP (WMMA) ---
  gemm_tiled<float, true, true><<<dim3(65536 / 64, 256 / 64), 128, 0, stream>>>(
      features, fc0h, fc_b0, H1, 65536, 512, 256);
  gemm_tiled<f16, true, true><<<dim3(65536 / 64, 64 / 64), 128, 0, stream>>>(
      H1, fc1h, fc_b1, H2, 65536, 256, 64);
  score_kernel<<<65536 / 256, 256, 0, stream>>>(H2, fc_w2, fc_b2, score, 65536);

  // --- top-k + ball query ---
  topk_kernel<<<16, 256, 0, stream>>>(score, scratch, topk, newxyz, points);
  ball_kernel<16><<<1024, 32, 0, stream>>>(points, newxyz, ball[0], R2s[0]);
  ball_kernel<32><<<1024, 32, 0, stream>>>(points, newxyz, ball[1], R2s[1]);
  ball_kernel<128><<<1024, 32, 0, stream>>>(points, newxyz, ball[2], R2s[2]);

  // --- SA branches (WMMA) ---
  sa_kernel<16, 32, 32, 64><<<1024, 128, 0, stream>>>(
      points, features, ball[0], newxyz, w0h[0], sb0[0], w1h[0], sb1[0],
      w2h[0], sb2[0], sa_out, OFFs[0]);
  sa_kernel<32, 64, 64, 128><<<1024, 128, 0, stream>>>(
      points, features, ball[1], newxyz, w0h[1], sb0[1], w1h[1], sb1[1],
      w2h[1], sb2[1], sa_out, OFFs[1]);
  sa_kernel<128, 64, 96, 128><<<1024, 128, 0, stream>>>(
      points, features, ball[2], newxyz, w0h[2], sb0[2], w1h[2], sb1[2],
      w2h[2], sb2[2], sa_out, OFFs[2]);

  // --- agg MLP (WMMA) ---
  gemm_tiled<float, true, true><<<dim3(1024 / 64, 256 / 64), 128, 0, stream>>>(
      sa_out, agg0h, agg_b0, aggH0, 1024, 320, 256);
  gemm_tiled<f16, true, true><<<dim3(1024 / 64, 256 / 64), 128, 0, stream>>>(
      aggH0, agg1h, agg_b1, aggH1, 1024, 256, 256);
  gemm_tiled<f16, false, false><<<dim3(1024 / 64, 512 / 64), 128, 0, stream>>>(
      aggH1, agg2h, aggb2p, aggout, 1024, 256, 512);

  // --- final concat ---
  assemble_kernel<<<(1024 * 512) / 256, 256, 0, stream>>>(aggout, score, points,
                                                          topk, out);
}
